// MoE_66314295050380
// MI455X (gfx1250) — matrix operand order repaired and checked by reading, not compiled, not run on hardware
//
#include <hip/hip_runtime.h>
#include <hip/hip_bf16.h>

// MoE forward for gfx1250 (MI455X): gathered grouped-GEMM, bf16 WMMA w/ fp32 acc,
// async global->LDS double-buffering in GEMM2.
// D=2048, H=2816, E=8, M=4096, top_k=2.
//
// Workspace layout (needs ~47.2 MB):
//   [0,32)        counts[8]            (int)
//   [32,64)       bases[8]             (int)
//   [256, +128K)  expert_tokens[E*M]   (int)
//   [+128K,+256K) expert_gates[E*M]    (float)
//   [...]         hbuf[(2M+128)*H]     (bf16)  ~46.9 MB

#define Dm 2048
#define Hm 2816
#define Em 8
#define Mm 4096
#define BM 128
#define KT 32
#define LP2 40   // GEMM2 LDS row pitch in bf16 (80 B: 16-B aligned for async b128)

typedef __attribute__((ext_vector_type(16))) __bf16 v16bf;
typedef __attribute__((ext_vector_type(8)))  float  v8f;

// A-matrix (16x32 bf16) per-lane K offset for VGPR dword v (holds K=k,k+1):
// lanes 0-15: V0..3 -> K=0..7, V4..7 -> K=16..23 ; lanes 16-31: +8
__device__ __forceinline__ int kOffA(int v, int khalf) {
  return (v < 4) ? (khalf * 8 + 2 * v) : (16 + khalf * 8 + 2 * (v - 4));
}
// B-matrix (32x16 bf16): same K order as A: lanes 0-15 V0..3 -> K=0..7, V4..7 -> K=16..23 ; lanes 16-31: +8
__device__ __forceinline__ int kOffB(int v, int khalf) {
  return (v < 4) ? (khalf * 8 + 2 * v) : (16 + khalf * 8 + 2 * (v - 4));
}

// 32-byte per-lane async copy global->LDS (CDNA5, ASYNCcnt-tracked).
// offset:16 advances BOTH dsaddr and memaddr per the ISA address formulas.
__device__ __forceinline__ void async_copy32(unsigned lds_off, const void* gp) {
  asm volatile("global_load_async_to_lds_b128 %0, %1, off\n\t"
               "global_load_async_to_lds_b128 %0, %1, off offset:16"
               :: "v"(lds_off), "v"(gp) : "memory");
}
__device__ __forceinline__ void wait_async0() {
  asm volatile("s_wait_asynccnt 0x0" ::: "memory");
}

// ---------------- kernel 1: init output with bias, zero counts ----------------
__global__ void moe_init_kernel(float* __restrict__ y,
                                const float* __restrict__ bias,
                                int* __restrict__ counts) {
  if (blockIdx.x == 0 && threadIdx.x < Em) counts[threadIdx.x] = 0;
  size_t idx = ((size_t)blockIdx.x * blockDim.x + threadIdx.x) * 4;
  if (idx < (size_t)Mm * Dm) {
    int d = (int)(idx % Dm);
    float4 b4 = *(const float4*)(bias + d);
    *(float4*)(y + idx) = b4;
  }
}

// ---------------- kernel 2: router (one wave32 per token) ----------------
__global__ void moe_router_kernel(const float* __restrict__ x,
                                  const float* __restrict__ rw,
                                  int* __restrict__ counts,
                                  int* __restrict__ toks,
                                  float* __restrict__ gates) {
  const int wave = threadIdx.x >> 5;
  const int lane = threadIdx.x & 31;
  const int t = blockIdx.x * (blockDim.x >> 5) + wave;
  if (t >= Mm) return;
  const float* xp = x + (size_t)t * Dm;
  float lg[Em];
#pragma unroll
  for (int e = 0; e < Em; ++e) {
    const float* wp = rw + (size_t)e * Dm;
    float s = 0.f;
    for (int j = lane; j < Dm; j += 32) s = fmaf(xp[j], wp[j], s);
#pragma unroll
    for (int off = 16; off > 0; off >>= 1) s += __shfl_xor(s, off, 32);
    lg[e] = s;
  }
  // softmax numerators (denominator cancels in top-2 renormalization)
  float mx = lg[0];
#pragma unroll
  for (int e = 1; e < Em; ++e) mx = fmaxf(mx, lg[e]);
  float p[Em];
#pragma unroll
  for (int e = 0; e < Em; ++e) p[e] = __expf(lg[e] - mx);
  // top-2 (first-index tie-break like lax.top_k)
  int i1 = 0;
#pragma unroll
  for (int e = 1; e < Em; ++e) if (p[e] > p[i1]) i1 = e;
  int i2 = (i1 == 0) ? 1 : 0;
#pragma unroll
  for (int e = 0; e < Em; ++e) if (e != i1 && p[e] > p[i2]) i2 = e;
  float g1 = p[i1], g2 = p[i2];
  float inv = 1.f / (g1 + g2);
  g1 *= inv; g2 *= inv;
  if (lane == 0) {
    int s1 = atomicAdd(&counts[i1], 1);
    toks[i1 * Mm + s1] = t; gates[i1 * Mm + s1] = g1;
    int s2 = atomicAdd(&counts[i2], 1);
    toks[i2 * Mm + s2] = t; gates[i2 * Mm + s2] = g2;
  }
}

// ---------------- kernel 3: exclusive prefix over 8 counts ----------------
__global__ void moe_prefix_kernel(const int* __restrict__ counts,
                                  int* __restrict__ bases) {
  if (threadIdx.x == 0) {
    int s = 0;
#pragma unroll
    for (int e = 0; e < Em; ++e) { bases[e] = s; s += counts[e]; }
  }
}

// ---------------- kernel 4: grouped GEMM1 + silu*mul -> hbuf (bf16) ----------------
// grid = (Hm/64, Mm/BM, Em), block = 256 (8 waves: 2 row-halves x 4 col-slices)
__global__ __launch_bounds__(256)
void moe_gemm1_kernel(const float* __restrict__ x,
                      const float* __restrict__ w1,
                      const int* __restrict__ counts,
                      const int* __restrict__ bases,
                      const int* __restrict__ toks,
                      __bf16* __restrict__ hbuf) {
  const int e = blockIdx.z;
  const int cnt = counts[e];
  const int rowTile = blockIdx.y;
  if (rowTile * BM >= cnt) return;          // uniform exit -> EXEC all-1 at WMMA
  const int colBase = blockIdx.x * 64;      // h columns

  __shared__ __bf16 As[BM][KT + 2];
  __shared__ int smTok[BM];

  const int tid = threadIdx.x;
  const int lane = tid & 31;
  const int wv = tid >> 5;
  const int wm = wv & 1;    // row half (64 rows)
  const int wn = wv >> 1;   // 16-col slice
  const int khalf = lane >> 4;

  if (tid < BM) {
    int lr = rowTile * BM + tid;
    smTok[tid] = (lr < cnt) ? toks[e * Mm + lr] : toks[e * Mm];
  }

  const float* w1e = w1 + (size_t)e * (2 * Hm) * Dm;
  const int nCol = colBase + wn * 16 + (lane & 15);
  const float* bGatePtr = w1e + (size_t)nCol * Dm;
  const float* bUpPtr   = w1e + (size_t)(Hm + nCol) * Dm;

  v8f accG[4] = {};
  v8f accU[4] = {};

  for (int k0 = 0; k0 < Dm; k0 += KT) {
    __syncthreads();
    { // cooperative A-tile load: 128 rows x 32 K, fp32 -> bf16 into LDS
      int row = tid >> 1;
      int ks  = (tid & 1) * 16;
      const float* xp = x + (size_t)smTok[row] * Dm + k0 + ks;
#pragma unroll
      for (int i = 0; i < 16; ++i) As[row][ks + i] = (__bf16)xp[i];
    }
    __syncthreads();

    // stream next weight rows toward L2
    if (k0 + KT < Dm) {
      __builtin_prefetch(bGatePtr + k0 + KT, 0, 1);
      __builtin_prefetch(bUpPtr   + k0 + KT, 0, 1);
    }

    // B fragments (gate & up) from global fp32 -> bf16
    v16bf bg, bu;
#pragma unroll
    for (int v = 0; v < 8; ++v) {
      int k = kOffB(v, khalf);
      float2 fg = *(const float2*)(bGatePtr + k0 + k);
      float2 fu = *(const float2*)(bUpPtr   + k0 + k);
      bg[2 * v] = (__bf16)fg.x; bg[2 * v + 1] = (__bf16)fg.y;
      bu[2 * v] = (__bf16)fu.x; bu[2 * v + 1] = (__bf16)fu.y;
    }

    // preload all A fragments, then the WMMA burst (scheduler can overlap)
    v16bf afr[4];
#pragma unroll
    for (int mt = 0; mt < 4; ++mt) {
      int row = wm * 64 + mt * 16 + (lane & 15);
#pragma unroll
      for (int v = 0; v < 8; ++v) {
        int k = kOffA(v, khalf);
        afr[mt][2 * v]     = As[row][k];
        afr[mt][2 * v + 1] = As[row][k + 1];
      }
    }
#pragma unroll
    for (int mt = 0; mt < 4; ++mt) {
      accG[mt] = __builtin_amdgcn_wmma_f32_16x16x32_bf16(
          false, afr[mt], false, bg, (short)0, accG[mt], false, false);
      accU[mt] = __builtin_amdgcn_wmma_f32_16x16x32_bf16(
          false, afr[mt], false, bu, (short)0, accU[mt], false, false);
    }
  }

  // epilogue: h = silu(gate) * up  -> bf16 hbuf (packed rows at bases[e])
  const int hBase = bases[e];
#pragma unroll
  for (int mt = 0; mt < 4; ++mt) {
#pragma unroll
    for (int r = 0; r < 8; ++r) {
      int rowLocal = wm * 64 + mt * 16 + khalf * 8 + r;
      int lr = rowTile * BM + rowLocal;
      if (lr < cnt) {
        float g = accG[mt][r];
        float u = accU[mt][r];
        float hv = (g / (1.f + __expf(-g))) * u;
        hbuf[(size_t)(hBase + lr) * Hm + nCol] = (__bf16)hv;
      }
    }
  }
}

// ---------------- kernel 5: grouped GEMM2 -> scaled atomic scatter to y ----------------
// grid = (Dm/64, Mm/BM, Em), block = 256
// A tile (bf16 hbuf rows) staged via GLOBAL_LOAD_ASYNC_TO_LDS_B128, double-buffered.
__global__ __launch_bounds__(256)
void moe_gemm2_kernel(const __bf16* __restrict__ hbuf,
                      const float* __restrict__ w2,
                      const int* __restrict__ counts,
                      const int* __restrict__ bases,
                      const int* __restrict__ toks,
                      const float* __restrict__ gates,
                      float* __restrict__ y) {
  const int e = blockIdx.z;
  const int cnt = counts[e];
  const int rowTile = blockIdx.y;
  if (rowTile * BM >= cnt) return;
  const int colBase = blockIdx.x * 64;      // output D columns

  __shared__ __bf16 As[2][BM][LP2];         // 80-B row pitch: 16-B aligned rows

  const int tid = threadIdx.x;
  const int lane = tid & 31;
  const int wv = tid >> 5;
  const int wm = wv & 1;
  const int wn = wv >> 1;
  const int khalf = lane >> 4;
  const int hBase = bases[e];

  const float* w2e = w2 + (size_t)e * Dm * Hm;
  const int nCol = colBase + wn * 16 + (lane & 15);
  const float* bPtr = w2e + (size_t)nCol * Hm;

  // per-thread async-copy source: 32 B (16 bf16) of one hbuf row
  const int cRow = tid >> 1;
  const int cKs  = (tid & 1) * 16;
  const __bf16* cSrc = hbuf + (size_t)(hBase + rowTile * BM + cRow) * Hm + cKs;

  v8f acc[4] = {};

  // prologue: bring tile k0=0 into buffer 0
  async_copy32((unsigned)(uintptr_t)&As[0][cRow][cKs], (const void*)cSrc);
  wait_async0();
  __syncthreads();

  int cur = 0;
  for (int k0 = 0; k0 < Hm; k0 += KT) {
    // issue next tile's async copy into the other buffer (overlaps with WMMAs)
    if (k0 + KT < Hm) {
      async_copy32((unsigned)(uintptr_t)&As[cur ^ 1][cRow][cKs],
                   (const void*)(cSrc + k0 + KT));
      __builtin_prefetch(bPtr + k0 + KT, 0, 1);
    }

    // B fragment from global fp32 -> bf16
    v16bf b;
#pragma unroll
    for (int v = 0; v < 8; ++v) {
      int k = kOffB(v, khalf);
      float2 f = *(const float2*)(bPtr + k0 + k);
      b[2 * v] = (__bf16)f.x; b[2 * v + 1] = (__bf16)f.y;
    }

    // preload A fragments from current buffer, then WMMA burst
    v16bf afr[4];
#pragma unroll
    for (int mt = 0; mt < 4; ++mt) {
      int row = wm * 64 + mt * 16 + (lane & 15);
#pragma unroll
      for (int v = 0; v < 8; ++v) {
        int k = kOffA(v, khalf);
        afr[mt][2 * v]     = As[cur][row][k];
        afr[mt][2 * v + 1] = As[cur][row][k + 1];
      }
    }
#pragma unroll
    for (int mt = 0; mt < 4; ++mt) {
      acc[mt] = __builtin_amdgcn_wmma_f32_16x16x32_bf16(
          false, afr[mt], false, b, (short)0, acc[mt], false, false);
    }

    // next buffer must be resident before anyone reads it next iteration
    wait_async0();
    __syncthreads();
    cur ^= 1;
  }

  // epilogue: y[tok, nCol] += gate * acc   (token appears in exactly 2 experts)
#pragma unroll
  for (int mt = 0; mt < 4; ++mt) {
#pragma unroll
    for (int r = 0; r < 8; ++r) {
      int rowLocal = wm * 64 + mt * 16 + khalf * 8 + r;
      int lr = rowTile * BM + rowLocal;
      if (lr < cnt) {
        int tok  = toks[e * Mm + lr];
        float gw = gates[e * Mm + lr];
        float val = acc[mt][r] * gw;
        __hip_atomic_fetch_add(&y[(size_t)tok * Dm + nCol], val,
                               __ATOMIC_RELAXED, __HIP_MEMORY_SCOPE_AGENT);
      }
    }
  }
}

// ---------------- launch ----------------
extern "C" void kernel_launch(void* const* d_in, const int* in_sizes, int n_in,
                              void* d_out, int out_size, void* d_ws, size_t ws_size,
                              hipStream_t stream) {
  const float* x    = (const float*)d_in[0];  // [M, D]
  const float* w1   = (const float*)d_in[1];  // [E, 2H, D]
  const float* w2   = (const float*)d_in[2];  // [E, D, H]
  const float* rw   = (const float*)d_in[3];  // [E, D]
  const float* bias = (const float*)d_in[4];  // [D]
  // d_in[5] = top_k (==2, hardcoded)
  float* y = (float*)d_out;                   // [M, D]

  char* ws = (char*)d_ws;
  int*    counts = (int*)(ws + 0);
  int*    bases  = (int*)(ws + 32);
  int*    toks   = (int*)(ws + 256);
  float*  gates  = (float*)(ws + 256 + (size_t)Em * Mm * 4);
  __bf16* hbuf   = (__bf16*)(ws + 256 + 2 * (size_t)Em * Mm * 4);
  // hbuf needs (2*Mm + 128) * Hm * 2 bytes ~= 46.9 MB; total ws ~= 47.2 MB

  {
    int blocks = (Mm * Dm / 4 + 255) / 256;
    moe_init_kernel<<<dim3(blocks), dim3(256), 0, stream>>>(y, bias, counts);
  }
  moe_router_kernel<<<dim3(Mm / 8), dim3(256), 0, stream>>>(x, rw, counts, toks, gates);
  moe_prefix_kernel<<<dim3(1), dim3(32), 0, stream>>>(counts, bases);
  moe_gemm1_kernel<<<dim3(Hm / 64, Mm / BM, Em), dim3(256), 0, stream>>>(
      x, w1, counts, bases, toks, hbuf);
  moe_gemm2_kernel<<<dim3(Dm / 64, Mm / BM, Em), dim3(256), 0, stream>>>(
      hbuf, w2, counts, bases, toks, gates, y);
}
